// GATLSTM_89240830476364
// MI455X (gfx1250) — matrix operand order repaired and checked by reading, not compiled, not run on hardware
//
#include <hip/hip_runtime.h>
#include <math.h>

// Problem constants (from reference)
#define NHEADS 2
#define FDIM   8          // HEADS * GAT_H = HEADS * LSTM_H = 8
#define NN_    128
#define BB_    16
#define TT_    128
#define FEAT_  1024       // LSTM hidden / input
#define GATES_ 4096       // 4 * FEAT
#define NTILE  4          // column tiles per wave in WMMA GEMM (16x64 strip)

typedef __attribute__((ext_vector_type(16))) _Float16 v16h;
typedef __attribute__((ext_vector_type(8)))  float    v8f;

__device__ __forceinline__ float lrelu02(float x){ return x > 0.f ? x : 0.2f * x; }
__device__ __forceinline__ float sigmoidf(float x){ return 1.f / (1.f + __expf(-x)); }

// float atomic max via integer ordering trick (init value must be -inf)
__device__ __forceinline__ void atomicMaxF(float* addr, float v){
    if (v >= 0.f) atomicMax((int*)addr, __float_as_int(v));
    else          atomicMin((unsigned int*)addr, __float_as_uint(v));
}

// ---------------------------------------------------------------- GAT kernels
__global__ void k_reset_accum(float* __restrict__ m, float* __restrict__ den,
                              float* __restrict__ acc, int N){
    int n = blockIdx.x * blockDim.x + threadIdx.x;
    if (n >= N) return;
    m[n*2+0] = -INFINITY; m[n*2+1] = -INFINITY;
    den[n*2+0] = 0.f;     den[n*2+1] = 0.f;
#pragma unroll
    for (int j = 0; j < FDIM; ++j) acc[n*FDIM + j] = 0.f;
}

// h = in @ W  ([N,D]x[D,8]); per-head attention logits a_src.h, a_dst.h
__global__ void k_gat_node(const float* __restrict__ in, int D,
                           const float* __restrict__ W,      // [D,8] row-major
                           const float* __restrict__ a_src,  // [2,4]
                           const float* __restrict__ a_dst,  // [2,4]
                           float* __restrict__ hpre, float* __restrict__ as_,
                           float* __restrict__ ad_, int N){
    int n = blockIdx.x * blockDim.x + threadIdx.x;
    if (n >= N) return;
    float h[FDIM];
#pragma unroll
    for (int j = 0; j < FDIM; ++j){
        float a = 0.f;
        for (int i = 0; i < D; ++i) a += in[n*D + i] * W[i*FDIM + j];
        h[j] = a; hpre[n*FDIM + j] = a;
    }
#pragma unroll
    for (int hd = 0; hd < NHEADS; ++hd){
        float s = 0.f, d = 0.f;
#pragma unroll
        for (int c = 0; c < 4; ++c){
            s += h[hd*4 + c] * a_src[hd*4 + c];
            d += h[hd*4 + c] * a_dst[hd*4 + c];
        }
        as_[n*2 + hd] = s; ad_[n*2 + hd] = d;
    }
}

__global__ void k_edge_max(const int* __restrict__ esrc, const int* __restrict__ edst,
                           int E, int N, const float* __restrict__ as_,
                           const float* __restrict__ ad_, float* __restrict__ m){
    int e = blockIdx.x * blockDim.x + threadIdx.x;
    if (e >= E + N) return;
    int s, d;
    if (e < E){ s = esrc[e]; d = edst[e]; } else { s = d = e - E; }   // self loops
#pragma unroll
    for (int hd = 0; hd < NHEADS; ++hd)
        atomicMaxF(&m[d*2 + hd], lrelu02(as_[s*2 + hd] + ad_[d*2 + hd]));
}

__global__ void k_edge_sum(const int* __restrict__ esrc, const int* __restrict__ edst,
                           int E, int N, const float* __restrict__ as_,
                           const float* __restrict__ ad_, const float* __restrict__ m,
                           float* __restrict__ den){
    int e = blockIdx.x * blockDim.x + threadIdx.x;
    if (e >= E + N) return;
    int s, d;
    if (e < E){ s = esrc[e]; d = edst[e]; } else { s = d = e - E; }
#pragma unroll
    for (int hd = 0; hd < NHEADS; ++hd){
        float mv = m[d*2 + hd]; if (!(mv > -1e30f)) mv = 0.f;
        float p = __expf(lrelu02(as_[s*2 + hd] + ad_[d*2 + hd]) - mv);
        atomicAdd(&den[d*2 + hd], p);
    }
}

__global__ void k_edge_scatter(const int* __restrict__ esrc, const int* __restrict__ edst,
                               int E, int N, const float* __restrict__ as_,
                               const float* __restrict__ ad_, const float* __restrict__ m,
                               const float* __restrict__ den, const float* __restrict__ hpre,
                               float* __restrict__ acc){
    int e = blockIdx.x * blockDim.x + threadIdx.x;
    if (e >= E + N) return;
    int s, d;
    if (e < E){ s = esrc[e]; d = edst[e]; } else { s = d = e - E; }
    float alpha[NHEADS];
#pragma unroll
    for (int hd = 0; hd < NHEADS; ++hd){
        float mv = m[d*2 + hd]; if (!(mv > -1e30f)) mv = 0.f;
        float p = __expf(lrelu02(as_[s*2 + hd] + ad_[d*2 + hd]) - mv);
        alpha[hd] = p / (den[d*2 + hd] + 1e-16f);
    }
#pragma unroll
    for (int j = 0; j < FDIM; ++j)
        atomicAdd(&acc[d*FDIM + j], hpre[s*FDIM + j] * alpha[j >> 2]);
}

__global__ void k_gat_finish1(const float* __restrict__ acc, const float* __restrict__ b,
                              float* __restrict__ out, int N){
    int n = blockIdx.x * blockDim.x + threadIdx.x;
    if (n >= N) return;
#pragma unroll
    for (int j = 0; j < FDIM; ++j){
        float v = acc[n*FDIM + j] + b[j];
        out[n*FDIM + j] = v > 0.f ? v : 0.f;
    }
}

// layer2 finish: relu(+b2) and pack into f16 LSTM sequence [B*T, FEAT]
__global__ void k_gat_finish2(const float* __restrict__ acc, const float* __restrict__ b,
                              _Float16* __restrict__ seq16, int N){
    int n = blockIdx.x * blockDim.x + threadIdx.x;
    if (n >= N) return;
    int g = n / NN_, s = n % NN_;                      // g = b*T + t
#pragma unroll
    for (int j = 0; j < FDIM; ++j){
        float v = acc[n*FDIM + j] + b[j];
        v = v > 0.f ? v : 0.f;
        seq16[(long long)g*FEAT_ + s*FDIM + j] = (_Float16)v;
    }
}

// ------------------------------------------------------------- LSTM (WMMA)
__global__ void k_cvt_f16(const float* __restrict__ src, _Float16* __restrict__ dst, int n){
    int i = blockIdx.x * blockDim.x + threadIdx.x;
    if (i < n) dst[i] = (_Float16)src[i];
}

__global__ void k_init_lstm(float* __restrict__ h32, float* __restrict__ c,
                            _Float16* __restrict__ h16){
    int i = blockIdx.x * blockDim.x + threadIdx.x;
    if (i >= BB_ * FEAT_) return;
    h32[i] = 0.f; c[i] = 0.f; h16[i] = (_Float16)0.f;
}

// A fragment (16x32, f16, row-major A with leading dim lda), per ISA §7.12.2:
// lanes 0-15: row M=lane, K = {0..7, 16..23}; lanes 16-31: row M=lane-16, K = {8..15, 24..31}
__device__ __forceinline__ v16h load_a16(const _Float16* __restrict__ A, int lda,
                                         int row0, int k0, int lane){
    const _Float16* p = A + (long long)(row0 + (lane & 15)) * lda + k0 + ((lane >> 4) << 3);
    union { float4 f4[2]; v16h v; } u;
    u.f4[0] = *(const float4*)(p);        // K = kh .. kh+7
    u.f4[1] = *(const float4*)(p + 16);   // K = 16+kh .. 16+kh+7
    return u.v;
}

// B fragment (32x16, f16) taken from row-major W[Nrows, ldw] so B[k][n] = W[n0+n][k0+k]:
// lanes 0-15: col N=lane, K = {0..7, 16..23}; lanes 16-31: col N=lane-16, K = {8..15, 24..31}
__device__ __forceinline__ v16h load_b16(const _Float16* __restrict__ W, int ldw,
                                         int n0, int k0, int lane){
    const _Float16* p = W + (long long)(n0 + (lane & 15)) * ldw + k0 + ((lane >> 4) << 3);
    union { float4 f4[2]; v16h v; } u;
    u.f4[0] = *(const float4*)(p);        // K = kh .. kh+7
    u.f4[1] = *(const float4*)(p + 16);   // K = 16+kh .. 16+kh+7
    return u.v;
}

// C = A(f16, [M,K]) @ W(f16, [Ncols,K])^T  (+ optional bias0+bias1 per column), f32 out.
// One wave computes a 16x(16*NTILE) strip: the A fragment is loaded ONCE per k-step and
// reused across NTILE independent accumulators -> ~1.6x fewer vector loads per WMMA and
// the 4 independent WMMAs cover the D->A/B RAW hazard window back-to-back.
__global__ void k_gemm_awt(const _Float16* __restrict__ A, const _Float16* __restrict__ W,
                           float* __restrict__ C, int K, int ldc,
                           const float* __restrict__ bias0, const float* __restrict__ bias1){
    int lane = threadIdx.x;
    int col0 = blockIdx.x * (16 * NTILE);
    int row0 = blockIdx.y * 16;
    v8f acc[NTILE];
#pragma unroll
    for (int t = 0; t < NTILE; ++t) acc[t] = (v8f){};
#pragma unroll 2
    for (int k0 = 0; k0 < K; k0 += 32){
        v16h a = load_a16(A, K, row0, k0, lane);
        v16h b0 = load_b16(W, K, col0 + 0*16, k0, lane);
        v16h b1 = load_b16(W, K, col0 + 1*16, k0, lane);
        v16h b2 = load_b16(W, K, col0 + 2*16, k0, lane);
        v16h b3 = load_b16(W, K, col0 + 3*16, k0, lane);
        acc[0] = __builtin_amdgcn_wmma_f32_16x16x32_f16(false, a, false, b0,
                                                        (short)0, acc[0], false, false);
        acc[1] = __builtin_amdgcn_wmma_f32_16x16x32_f16(false, a, false, b1,
                                                        (short)0, acc[1], false, false);
        acc[2] = __builtin_amdgcn_wmma_f32_16x16x32_f16(false, a, false, b2,
                                                        (short)0, acc[2], false, false);
        acc[3] = __builtin_amdgcn_wmma_f32_16x16x32_f16(false, a, false, b3,
                                                        (short)0, acc[3], false, false);
    }
    // C/D layout: VGPR r -> row = r (+8 for lanes 16-31), col = lane&15
    int rbase = row0 + ((lane >> 4) << 3);
#pragma unroll
    for (int t = 0; t < NTILE; ++t){
        int col = col0 + t*16 + (lane & 15);
        float bs = 0.f;
        if (bias0) bs += bias0[col];
        if (bias1) bs += bias1[col];
#pragma unroll
        for (int r = 0; r < 8; ++r)
            C[(long long)(rbase + r) * ldc + col] = acc[t][r] + bs;
    }
}

// Per-timestep gate nonlinearity + state update (biases already folded into gx)
__global__ void k_lstm_point(const float* __restrict__ gx, const float* __restrict__ gh,
                             float* __restrict__ c, float* __restrict__ h32,
                             _Float16* __restrict__ h16, int t){
    int idx = blockIdx.x * blockDim.x + threadIdx.x;
    if (idx >= BB_ * FEAT_) return;
    int b = idx >> 10, j = idx & 1023;
    long long rx = (long long)(b * TT_ + t) * GATES_;
    long long rh = (long long)b * GATES_;
    float gi = gx[rx + j]             + gh[rh + j];
    float gf = gx[rx + FEAT_ + j]     + gh[rh + FEAT_ + j];
    float gg = gx[rx + 2*FEAT_ + j]   + gh[rh + 2*FEAT_ + j];
    float go = gx[rx + 3*FEAT_ + j]   + gh[rh + 3*FEAT_ + j];
    float cn = sigmoidf(gf) * c[idx] + sigmoidf(gi) * tanhf(gg);
    float hn = sigmoidf(go) * tanhf(cn);
    c[idx] = cn; h32[idx] = hn; h16[idx] = (_Float16)hn;
}

// out[b][o] = h[b] . W_fc[o] + b_fc[o]   (48 outputs)
__global__ void k_fc(const float* __restrict__ h, const float* __restrict__ Wfc,
                     const float* __restrict__ bfc, float* __restrict__ out){
    int o = blockIdx.x;                 // 0..47
    int b = o / 3, oc = o % 3;
    __shared__ float red[256];
    float a = 0.f;
    for (int k = threadIdx.x; k < FEAT_; k += 256)
        a += h[b*FEAT_ + k] * Wfc[oc*FEAT_ + k];
    red[threadIdx.x] = a; __syncthreads();
    for (int s = 128; s > 0; s >>= 1){
        if (threadIdx.x < s) red[threadIdx.x] += red[threadIdx.x + s];
        __syncthreads();
    }
    if (threadIdx.x == 0) out[b*3 + oc] = red[0] + bfc[oc];
}

// ---------------------------------------------------------------- launcher
extern "C" void kernel_launch(void* const* d_in, const int* in_sizes, int n_in,
                              void* d_out, int out_size, void* d_ws, size_t ws_size,
                              hipStream_t stream){
    const float* x      = (const float*)d_in[0];
    const int*   eidx   = (const int*)  d_in[1];
    const float* W1     = (const float*)d_in[2];
    const float* a_src1 = (const float*)d_in[3];
    const float* a_dst1 = (const float*)d_in[4];
    const float* b1     = (const float*)d_in[5];
    const float* W2     = (const float*)d_in[6];
    const float* a_src2 = (const float*)d_in[7];
    const float* a_dst2 = (const float*)d_in[8];
    const float* b2     = (const float*)d_in[9];
    const float* W_ih   = (const float*)d_in[10];
    const float* W_hh   = (const float*)d_in[11];
    const float* b_ih   = (const float*)d_in[12];
    const float* b_hh   = (const float*)d_in[13];
    const float* W_fc   = (const float*)d_in[14];
    const float* b_fc   = (const float*)d_in[15];

    const int N = in_sizes[0];            // IN_DIM == 1 -> node count
    const int E = in_sizes[1] / 2;
    const int* esrc = eidx;
    const int* edst = eidx + E;

    // workspace carve-out
    char* base = (char*)d_ws; size_t off = 0;
    auto alloc = [&](size_t bytes)->void*{
        void* p = base + off; off = (off + bytes + 255) & ~(size_t)255; return p;
    };
    float*    hpre   = (float*)   alloc((size_t)N * FDIM * 4);
    float*    feat1  = (float*)   alloc((size_t)N * FDIM * 4);
    float*    as_    = (float*)   alloc((size_t)N * 2 * 4);
    float*    ad_    = (float*)   alloc((size_t)N * 2 * 4);
    float*    mbuf   = (float*)   alloc((size_t)N * 2 * 4);
    float*    den    = (float*)   alloc((size_t)N * 2 * 4);
    float*    acc    = (float*)   alloc((size_t)N * FDIM * 4);
    _Float16* seq16  = (_Float16*)alloc((size_t)BB_*TT_*FEAT_ * 2);
    _Float16* wih16  = (_Float16*)alloc((size_t)GATES_*FEAT_ * 2);
    _Float16* whh16  = (_Float16*)alloc((size_t)GATES_*FEAT_ * 2);
    float*    gatesx = (float*)   alloc((size_t)BB_*TT_*GATES_ * 4);
    float*    gh     = (float*)   alloc((size_t)BB_*GATES_ * 4);
    float*    h32    = (float*)   alloc((size_t)BB_*FEAT_ * 4);
    float*    cst    = (float*)   alloc((size_t)BB_*FEAT_ * 4);
    _Float16* h16    = (_Float16*)alloc((size_t)BB_*FEAT_ * 2);

    const int nb = (N + 255) / 256;
    const int eb = (E + N + 255) / 256;

    // ---- GAT layer 1
    k_reset_accum<<<nb, 256, 0, stream>>>(mbuf, den, acc, N);
    k_gat_node   <<<nb, 256, 0, stream>>>(x, 1, W1, a_src1, a_dst1, hpre, as_, ad_, N);
    k_edge_max   <<<eb, 256, 0, stream>>>(esrc, edst, E, N, as_, ad_, mbuf);
    k_edge_sum   <<<eb, 256, 0, stream>>>(esrc, edst, E, N, as_, ad_, mbuf, den);
    k_edge_scatter<<<eb,256, 0, stream>>>(esrc, edst, E, N, as_, ad_, mbuf, den, hpre, acc);
    k_gat_finish1<<<nb, 256, 0, stream>>>(acc, b1, feat1, N);

    // ---- GAT layer 2 (reuse accumulators)
    k_reset_accum<<<nb, 256, 0, stream>>>(mbuf, den, acc, N);
    k_gat_node   <<<nb, 256, 0, stream>>>(feat1, FDIM, W2, a_src2, a_dst2, hpre, as_, ad_, N);
    k_edge_max   <<<eb, 256, 0, stream>>>(esrc, edst, E, N, as_, ad_, mbuf);
    k_edge_sum   <<<eb, 256, 0, stream>>>(esrc, edst, E, N, as_, ad_, mbuf, den);
    k_edge_scatter<<<eb,256, 0, stream>>>(esrc, edst, E, N, as_, ad_, mbuf, den, hpre, acc);
    k_gat_finish2<<<nb, 256, 0, stream>>>(acc, b2, seq16, N);

    // ---- LSTM prep: f16 weights + zero state
    const int wcnt = GATES_ * FEAT_;
    k_cvt_f16<<<(wcnt + 255)/256, 256, 0, stream>>>(W_ih, wih16, wcnt);
    k_cvt_f16<<<(wcnt + 255)/256, 256, 0, stream>>>(W_hh, whh16, wcnt);
    k_init_lstm<<<(BB_*FEAT_ + 255)/256, 256, 0, stream>>>(h32, cst, h16);

    // ---- hoisted input GEMM: gates_x = seq @ W_ih^T + (b_ih + b_hh)   [2048 x 4096]
    {
        dim3 g(GATES_/(16*NTILE), (BB_*TT_)/16);     // 64 x 128 single-wave strips
        k_gemm_awt<<<g, 32, 0, stream>>>(seq16, wih16, gatesx, FEAT_, GATES_, b_ih, b_hh);
    }

    // ---- recurrence: 128 dependent steps, W_hh(f16, 8MB) stays L2-resident
    for (int t = 0; t < TT_; ++t){
        dim3 g(GATES_/(16*NTILE), BB_/16);           // 64 x 1 strips
        k_gemm_awt<<<g, 32, 0, stream>>>(h16, whh16, gh, FEAT_, GATES_, nullptr, nullptr);
        k_lstm_point<<<(BB_*FEAT_ + 255)/256, 256, 0, stream>>>(gatesx, gh, cst, h32, h16, t);
    }

    // ---- readout
    k_fc<<<BB_ * 3, 256, 0, stream>>>(h32, W_fc, b_fc, (float*)d_out);
}
